// MLP_difs_softmax_22625887715773
// MI455X (gfx1250) — matrix operand order repaired and checked by reading, not compiled, not run on hardware
//
#include <hip/hip_runtime.h>
#include <hip/hip_bf16.h>
#include <math.h>

typedef __attribute__((ext_vector_type(16))) _Float16 v16h;
typedef __attribute__((ext_vector_type(8)))  float    v8f;

// ---------------------------------------------------------------------------
// Kernel 0: zero a float buffer
// ---------------------------------------------------------------------------
__global__ void zero_kernel(float* __restrict__ p, int n) {
    int i = blockIdx.x * blockDim.x + threadIdx.x;
    if (i < n) p[i] = 0.f;
}

// ---------------------------------------------------------------------------
// Kernel 1: edge MLP via WMMA, M-blocked x2 for B-fragment reuse.
//   One wave = two tiles of 16 edges (M=32).  A = dif tiles [16 x 128] (f16),
//   B = W1 [128 x 128] (f16, staged in LDS in fragment order).  Each B
//   fragment (2x ds_load_b128/lane) feeds TWO v_wmma_f32_16x16x32_f16.
//   hidden never materialized: layer-2 dot with W2 fused in-register.
//   Writes sigmoid(att) to att[] and atomically accumulates denom[src].
// ---------------------------------------------------------------------------
__global__ __launch_bounds__(256) void edge_mlp_kernel(
    const float* __restrict__ h,
    const int*   __restrict__ src,
    const int*   __restrict__ dst,
    const float* __restrict__ W1,
    const float* __restrict__ b1,
    const float* __restrict__ W2,
    const float* __restrict__ b2,
    float* __restrict__ att,
    float* __restrict__ denom,
    int nTiles, int nPairs)
{
    // W1 fragments: [ntile(8)][kchunk(4)][lane(32)][16 halves] = 16384 halves = 32KB
    __shared__ __align__(32) _Float16 w1f[8 * 4 * 32 * 16];
    __shared__ float b1s[128];
    __shared__ float w2s[128];
    __shared__ float b2s;

    const int t = threadIdx.x;

    // Stage W1 into LDS, pre-swizzled to the 16-bit B-matrix (KxN, K=32) layout:
    //   lane L holds column N = L%16; kg = L/16; half hh -> K = 32*c + 8*kg + (hh&7) + 16*(hh>>3)
    #pragma unroll
    for (int i = 0; i < 64; ++i) {
        int idx  = i * 256 + t;              // 0..16383
        int hh   = idx & 15;
        int lane = (idx >> 4) & 31;
        int c    = (idx >> 9) & 3;
        int nt   = (idx >> 11) & 7;
        int kg   = lane >> 4;
        int N    = lane & 15;
        int K    = 32 * c + 8 * kg + (hh & 7) + 16 * (hh >> 3);
        int n    = 16 * nt + N;
        w1f[idx] = (_Float16)W1[K * 128 + n];
    }
    if (t < 128) { b1s[t] = b1[t]; w2s[t] = W2[t]; }
    if (t == 0)  { b2s = b2[0]; }
    __syncthreads();

    const int wave = t >> 5;
    const int lane = t & 31;
    const int pair = blockIdx.x * 8 + wave;
    if (pair >= nPairs) return;              // whole-wave uniform exit (EXEC stays full)

    const int tile0 = pair * 2;
    const int tile1 = pair * 2 + 1;
    const bool has1 = (tile1 < nTiles);

    const int m  = lane & 15;                // A-matrix row / C-matrix column index
    const int kg = lane >> 4;

    // ---- Build A fragments for both tiles: dif = h[src] - h[dst], f16 ------
    const int e0 = tile0 * 16 + m;
    const int e1 = has1 ? (tile1 * 16 + m) : e0;   // clamp: keep gathers in-bounds
    const int s0 = src[e0], d0 = dst[e0];
    const int s1 = src[e1], d1 = dst[e1];
    const float* __restrict__ hs0 = h + (size_t)s0 * 128;
    const float* __restrict__ hd0 = h + (size_t)d0 * 128;
    const float* __restrict__ hs1 = h + (size_t)s1 * 128;
    const float* __restrict__ hd1 = h + (size_t)d1 * 128;

    v16h a0[4], a1[4];
    #pragma unroll
    for (int c = 0; c < 4; ++c) {
        const int b0 = 32 * c + 8 * kg;      // ISA 16-bit A 16x32 layout
        #pragma unroll
        for (int j = 0; j < 8; ++j) {
            a0[c][j]     = (_Float16)(hs0[b0 + j]      - hd0[b0 + j]);
            a0[c][j + 8] = (_Float16)(hs0[b0 + 16 + j] - hd0[b0 + 16 + j]);
            a1[c][j]     = (_Float16)(hs1[b0 + j]      - hd1[b0 + j]);
            a1[c][j + 8] = (_Float16)(hs1[b0 + 16 + j] - hd1[b0 + 16 + j]);
        }
    }

    // ---- GEMM1 (+bias+relu) fused with GEMM2 dot against W2 ----------------
    float att0[8], att1[8];
    #pragma unroll
    for (int r = 0; r < 8; ++r) { att0[r] = 0.f; att1[r] = 0.f; }

    #pragma unroll
    for (int nt = 0; nt < 8; ++nt) {
        v8f acc0 = {};
        v8f acc1 = {};
        #pragma unroll
        for (int c = 0; c < 4; ++c) {
            const v16h bfrag = *(const v16h*)&w1f[((nt * 4 + c) * 32 + lane) * 16];
            acc0 = __builtin_amdgcn_wmma_f32_16x16x32_f16(
                false, a0[c], false, bfrag, (short)0, acc0, false, false);
            acc1 = __builtin_amdgcn_wmma_f32_16x16x32_f16(
                false, a1[c], false, bfrag, (short)0, acc1, false, false);
        }
        const int   n   = nt * 16 + m;       // C layout: this lane's column
        const float bb  = b1s[n];
        const float w2v = w2s[n];
        #pragma unroll
        for (int r = 0; r < 8; ++r) {
            float h0 = acc0[r] + bb;  h0 = h0 > 0.f ? h0 : 0.f;
            float h1 = acc1[r] + bb;  h1 = h1 > 0.f ? h1 : 0.f;
            att0[r] += h0 * w2v;
            att1[r] += h1 * w2v;
        }
    }

    // ---- Reduce over the 16 columns held by each half-wave ------------------
    #pragma unroll
    for (int r = 0; r < 8; ++r) {
        float v0 = att0[r];
        v0 += __shfl_xor(v0, 1); v0 += __shfl_xor(v0, 2);
        v0 += __shfl_xor(v0, 4); v0 += __shfl_xor(v0, 8);
        att0[r] = v0;
        float v1 = att1[r];
        v1 += __shfl_xor(v1, 1); v1 += __shfl_xor(v1, 2);
        v1 += __shfl_xor(v1, 4); v1 += __shfl_xor(v1, 8);
        att1[r] = v1;
    }

    // C layout: VGPR r holds row M = r + 8*kg.  Lane 0 finishes rows 0..7,
    // lane 16 finishes rows 8..15 (per tile).
    if ((lane & 15) == 0) {
        const int rowbase = kg * 8;
        #pragma unroll
        for (int r = 0; r < 8; ++r) {
            {
                const int ee = tile0 * 16 + rowbase + r;
                float av = att0[r] + b2s;
                av = av > 0.f ? av : 0.f;                    // relu
                const float sg = 1.f / (1.f + __expf(-av));  // sigmoid
                att[ee] = sg;
                atomicAdd(&denom[src[ee]], sg);
            }
            if (has1) {
                const int ee = tile1 * 16 + rowbase + r;
                float av = att1[r] + b2s;
                av = av > 0.f ? av : 0.f;
                const float sg = 1.f / (1.f + __expf(-av));
                att[ee] = sg;
                atomicAdd(&denom[src[ee]], sg);
            }
        }
    }
}

// ---------------------------------------------------------------------------
// Kernel 1b: scalar tail for edges beyond the last full 16-tile (rarely used)
// ---------------------------------------------------------------------------
__global__ void edge_mlp_tail_kernel(
    const float* __restrict__ h, const int* __restrict__ src, const int* __restrict__ dst,
    const float* __restrict__ W1, const float* __restrict__ b1,
    const float* __restrict__ W2, const float* __restrict__ b2,
    float* __restrict__ att, float* __restrict__ denom, int e0, int nEdges)
{
    const int wv   = (blockIdx.x * blockDim.x + threadIdx.x) >> 5;
    const int lane = threadIdx.x & 31;
    const int e    = e0 + wv;
    if (e >= nEdges) return;
    const int s = src[e], d = dst[e];
    const float* hs = h + (size_t)s * 128;
    const float* hd = h + (size_t)d * 128;
    float part = 0.f;
    for (int nb = 0; nb < 4; ++nb) {
        const int n = nb * 32 + lane;
        float acc = b1[n];
        for (int k = 0; k < 128; ++k)
            acc += (hs[k] - hd[k]) * W1[k * 128 + n];
        acc = acc > 0.f ? acc : 0.f;
        part += acc * W2[n];
    }
    for (int msk = 1; msk < 32; msk <<= 1) part += __shfl_xor(part, msk);
    if (lane == 0) {
        float av = part + b2[0];
        av = av > 0.f ? av : 0.f;
        const float sg = 1.f / (1.f + __expf(-av));
        att[e] = sg;
        atomicAdd(&denom[s], sg);
    }
}

// ---------------------------------------------------------------------------
// Kernel 2: normalize + scatter.  One wave per edge, float4 per lane.
//   out[dst] += (att[e] / denom[src[e]]) * h[src[e]]
// ---------------------------------------------------------------------------
__global__ __launch_bounds__(256) void scatter_kernel(
    const float* __restrict__ h,
    const int*   __restrict__ src,
    const int*   __restrict__ dst,
    const float* __restrict__ att,
    const float* __restrict__ denom,
    float* __restrict__ out,
    int nEdges)
{
    const int wv   = (blockIdx.x * blockDim.x + threadIdx.x) >> 5;
    const int lane = threadIdx.x & 31;
    if (wv >= nEdges) return;
    const int s = src[wv];
    const int d = dst[wv];
    const float w = att[wv] / denom[s];
    const float4 hv = ((const float4*)(h + (size_t)s * 128))[lane];
    float* op = out + (size_t)d * 128 + lane * 4;
    atomicAdd(op + 0, w * hv.x);
    atomicAdd(op + 1, w * hv.y);
    atomicAdd(op + 2, w * hv.z);
    atomicAdd(op + 3, w * hv.w);
}

// ---------------------------------------------------------------------------
extern "C" void kernel_launch(void* const* d_in, const int* in_sizes, int n_in,
                              void* d_out, int out_size, void* d_ws, size_t ws_size,
                              hipStream_t stream) {
    const float* h   = (const float*)d_in[0];
    const int*   src = (const int*)  d_in[1];
    const int*   dst = (const int*)  d_in[2];
    const float* W1  = (const float*)d_in[3];
    const float* b1  = (const float*)d_in[4];
    const float* W2  = (const float*)d_in[5];
    const float* b2  = (const float*)d_in[6];
    float* out = (float*)d_out;

    const int nNodes = in_sizes[0] / 128;
    const int nEdges = in_sizes[1];

    float* att   = (float*)d_ws;          // [nEdges]
    float* denom = att + nEdges;          // [nNodes]

    // Zero output + denom (harness poisons once; atomics need clean zeros).
    const int nOut = nNodes * 128;
    zero_kernel<<<(nOut + 255) / 256, 256, 0, stream>>>(out, nOut);
    zero_kernel<<<(nNodes + 255) / 256, 256, 0, stream>>>(denom, nNodes);

    // Edge MLP (WMMA) over full 16-edge tiles, 2 tiles per wave; scalar tail.
    const int nTiles = nEdges / 16;
    const int nPairs = (nTiles + 1) / 2;
    if (nPairs > 0) {
        edge_mlp_kernel<<<(nPairs + 7) / 8, 256, 0, stream>>>(
            h, src, dst, W1, b1, W2, b2, att, denom, nTiles, nPairs);
    }
    const int rem = nEdges - nTiles * 16;
    if (rem > 0) {
        edge_mlp_tail_kernel<<<(rem * 32 + 255) / 256, 256, 0, stream>>>(
            h, src, dst, W1, b1, W2, b2, att, denom, nTiles * 16, nEdges);
    }

    // Normalize + scatter-add messages.
    const long long scatterThreads = (long long)nEdges * 32;
    scatter_kernel<<<(int)((scatterThreads + 255) / 256), 256, 0, stream>>>(
        h, src, dst, att, denom, out, nEdges);
}
